// WiKG_90400471646498
// MI455X (gfx1250) — matrix operand order repaired and checked by reading, not compiled, not run on hardware
//
#include <hip/hip_runtime.h>
#include <stdint.h>

// ---------------- problem constants (match reference) ----------------
#define N_NODES   8000
#define D_IN      512
#define H_HEADS   3
#define C_CH      512
#define HC        1536            // H*C
#define K_NBR     6
#define N_EDGES   (N_NODES * (K_NBR + 1))   // 56000 (knn + self loops)
#define NEG_SLOPE 0.2f

// ---------------- types ----------------
typedef __attribute__((ext_vector_type(16))) __bf16       v16bf;
typedef __attribute__((ext_vector_type(8)))  float        v8f;
typedef __attribute__((ext_vector_type(8)))  unsigned int v8u;
typedef __attribute__((ext_vector_type(4)))  unsigned int u32x4;
typedef __attribute__((ext_vector_type(8)))  unsigned int u32x8;

static __device__ inline unsigned short f32_to_bf16_rne(float f) {
  unsigned u = __float_as_uint(f);
  unsigned r = 0x7FFFu + ((u >> 16) & 1u);
  return (unsigned short)((u + r) >> 16);
}
static __device__ inline float bf16_to_f32(unsigned short h) {
  return __uint_as_float(((unsigned)h) << 16);
}

union FragU { v8u u; v16bf b; };

// assemble a 16-element bf16 fragment from two 16B LDS chunks
static __device__ inline v16bf load_frag16(const unsigned short* p0,
                                           const unsigned short* p1) {
  uint4 a = *reinterpret_cast<const uint4*>(p0);
  uint4 b = *reinterpret_cast<const uint4*>(p1);
  FragU f;
  f.u[0] = a.x; f.u[1] = a.y; f.u[2] = a.z; f.u[3] = a.w;
  f.u[4] = b.x; f.u[5] = b.y; f.u[6] = b.z; f.u[7] = b.w;
  return f.b;
}

// ---------------- TDM: async 2-D bf16 tile load, global -> LDS ----------------
// tile_d1 rows x tile_d0 elements (2B each), row stride = stride_elems in global.
// LDS rows padded by 4 DWORDs every 16 DWORDs (pad_interval=3, pad_amount=3)
// -> LDS row stride = 64B + 16B = 80B, matching LDT=40 halfs (bank-conflict-free).
static __device__ inline void tdm_load_2d(unsigned lds_addr, const void* gptr,
                                          unsigned tile_d0, unsigned tile_d1,
                                          unsigned long long stride_elems)
{
  unsigned long long ga = (unsigned long long)(size_t)gptr;
  u32x4 g0;
  g0[0] = 1u;                                   // count=1, is_restore=0, gather off
  g0[1] = lds_addr;                             // lds_addr [63:32]
  g0[2] = (unsigned)(ga & 0xFFFFFFFFu);         // global_addr [95:64]
  g0[3] = (unsigned)((ga >> 32) & 0x1FFFFFFu)   // global_addr [120:96]
        | (2u << 30);                           // type=2 ("image")

  const unsigned long long td0 = 0x7FFFFFFFull; // huge tensor dims: no OOB clip
  const unsigned long long td1 = 0x7FFFFFFFull;
  u32x8 g1;
  g1[0] = (1u << 16)     // data_size = 2 bytes
        | (1u << 20)     // pad_enable
        | (3u << 22)     // pad_interval: 16 DWORDs
        | (3u << 25);    // pad_amount:   4 DWORDs
  g1[1] = (unsigned)((td0 & 0xFFFFull) << 16);                         // [63:48]=td0 lo16
  g1[2] = (unsigned)((td0 >> 16) & 0xFFFFull)
        | (unsigned)((td1 & 0xFFFFull) << 16);                          // td0 hi16 | td1 lo16
  g1[3] = (unsigned)((td1 >> 16) & 0xFFFFull) | (tile_d0 << 16);        // td1 hi16 | tile_dim0
  g1[4] = tile_d1 & 0xFFFFu;                                            // tile_dim1, tile_dim2=0
  g1[5] = (unsigned)(stride_elems & 0xFFFFFFFFull);                     // dim0_stride lo32
  g1[6] = (unsigned)((stride_elems >> 32) & 0xFFFFull);                 // dim0_stride hi16
  g1[7] = 0u;
  asm volatile("tensor_load_to_lds %0, %1" :: "s"(g0), "s"(g1) : "memory");
}

// ---------------- preconvert: fp32 -> bf16 hi/lo split ----------------
__global__ void cvt_split_kernel(const float* __restrict__ src,
                                 unsigned short* __restrict__ hi,
                                 unsigned short* __restrict__ lo, int n4)
{
  int g = blockIdx.x * blockDim.x + threadIdx.x;
  if (g >= n4) return;
  float4 f = reinterpret_cast<const float4*>(src)[g];
  ushort4 h, l;
  h.x = f32_to_bf16_rne(f.x); l.x = f32_to_bf16_rne(f.x - bf16_to_f32(h.x));
  h.y = f32_to_bf16_rne(f.y); l.y = f32_to_bf16_rne(f.y - bf16_to_f32(h.y));
  h.z = f32_to_bf16_rne(f.z); l.z = f32_to_bf16_rne(f.z - bf16_to_f32(h.z));
  h.w = f32_to_bf16_rne(f.w); l.w = f32_to_bf16_rne(f.w - bf16_to_f32(h.w));
  reinterpret_cast<ushort4*>(hi)[g] = h;
  reinterpret_cast<ushort4*>(lo)[g] = l;
}

// W (K x N fp32, K=D_IN rows, N=HC cols) -> transposed N x K bf16 hi/lo
__global__ void cvt_wT_kernel(const float* __restrict__ W,
                              unsigned short* __restrict__ hiT,
                              unsigned short* __restrict__ loT)
{
  int g = blockIdx.x * blockDim.x + threadIdx.x;   // g = n*K + k
  if (g >= HC * D_IN) return;
  int n = g / D_IN, k = g - n * D_IN;
  float f = W[(size_t)k * HC + n];
  unsigned short h = f32_to_bf16_rne(f);
  hiT[g] = h;
  loT[g] = f32_to_bf16_rne(f - bf16_to_f32(h));
}

// ---------------- split-bf16 WMMA GEMM, TDM-staged, double-buffered ----------------
// A: M x K bf16 (hi/lo), Bt: N x K bf16 (hi/lo, pre-transposed), Y = A*B + bias (fp32)
#define TM  64
#define TN  128
#define TK  32
#define LDT 40                         // padded LDS row stride in halfs (80B)
#define TILE_A_H (TM * LDT)            // 2560 halfs
#define TILE_B_H (TN * LDT)            // 5120 halfs
#define BUF_H    (2 * TILE_A_H + 2 * TILE_B_H)   // 15360 halfs / buffer

__global__ __launch_bounds__(256) void gemm_bias_bf16split_tdm(
    const unsigned short* __restrict__ Ahi_g, const unsigned short* __restrict__ Alo_g,
    const unsigned short* __restrict__ Bhi_g, const unsigned short* __restrict__ Blo_g,
    const float* __restrict__ bias, float* __restrict__ Y,
    int M, int K, int N)
{
  __shared__ __align__(16) unsigned short smem[2 * BUF_H];   // 60 KB

  const int tid  = threadIdx.x;
  const int bm   = blockIdx.x * TM;
  const int bn   = blockIdx.y * TN;
  const int wave = tid >> 5;
  const int lane = tid & 31;
  const int wm   = wave >> 2;   // 0..1
  const int wn   = wave & 3;    // 0..3

  const unsigned ldsBase = (unsigned)(size_t)(&smem[0]);  // flat low 32b = LDS offset

  auto issue = [&](int buf, int k0) {
    unsigned base = ldsBase + (unsigned)(buf * (BUF_H * 2));
    tdm_load_2d(base,                               Ahi_g + (size_t)bm * K + k0, TK, TM, (unsigned long long)K);
    tdm_load_2d(base + TILE_A_H * 2,                Alo_g + (size_t)bm * K + k0, TK, TM, (unsigned long long)K);
    tdm_load_2d(base + 2 * TILE_A_H * 2,            Bhi_g + (size_t)bn * K + k0, TK, TN, (unsigned long long)K);
    tdm_load_2d(base + (2 * TILE_A_H + TILE_B_H) * 2, Blo_g + (size_t)bn * K + k0, TK, TN, (unsigned long long)K);
  };

  const v8f vzero = {0.f, 0.f, 0.f, 0.f, 0.f, 0.f, 0.f, 0.f};
  v8f acc[2][2];
  for (int i = 0; i < 2; ++i)
    for (int j = 0; j < 2; ++j)
      acc[i][j] = vzero;

  // A fragment: lanes 0-15 rows m, K 0..7 & 16..23; lanes 16-31 K 8..15 & 24..31
  const int aRow = wm * 32 + (lane & 15);
  const int ka   = (lane >> 4) * 8;
  // B fragment: lanes 0-15 col n, K 0..7 & 16..23; lanes 16-31 K 8..15 & 24..31
  const int bCol = wn * 32 + (lane & 15);
  const int kb   = (lane >> 4) * 8;

  if (tid == 0) issue(0, 0);
  __builtin_amdgcn_s_wait_tensorcnt(0);
  __syncthreads();

  const int nIter = K / TK;
  for (int it = 0; it < nIter; ++it) {
    const int cur = it & 1;
    if (tid == 0 && (it + 1) < nIter) issue(cur ^ 1, (it + 1) * TK);   // overlap DMA

    const unsigned short* sA_hi = smem + cur * BUF_H;
    const unsigned short* sA_lo = sA_hi + TILE_A_H;
    const unsigned short* sB_hi = sA_hi + 2 * TILE_A_H;
    const unsigned short* sB_lo = sB_hi + TILE_B_H;

    v16bf aH[2], aL[2], bH[2], bL[2];
    #pragma unroll
    for (int i = 0; i < 2; ++i) {
      const unsigned short* ra = sA_hi + (aRow + i * 16) * LDT + ka;
      const unsigned short* la = sA_lo + (aRow + i * 16) * LDT + ka;
      const unsigned short* rb = sB_hi + (bCol + i * 16) * LDT + kb;
      const unsigned short* lb = sB_lo + (bCol + i * 16) * LDT + kb;
      aH[i] = load_frag16(ra, ra + 16);
      aL[i] = load_frag16(la, la + 16);
      bH[i] = load_frag16(rb, rb + 16);
      bL[i] = load_frag16(lb, lb + 16);
    }

    // 3-term split product: hi*hi + hi*lo + lo*hi  (~fp32 accuracy)
    #pragma unroll
    for (int i = 0; i < 2; ++i) {
      #pragma unroll
      for (int j = 0; j < 2; ++j) {
        acc[i][j] = __builtin_amdgcn_wmma_f32_16x16x32_bf16(
            false, aH[i], false, bH[j], (short)0, acc[i][j], false, false);
        acc[i][j] = __builtin_amdgcn_wmma_f32_16x16x32_bf16(
            false, aH[i], false, bL[j], (short)0, acc[i][j], false, false);
        acc[i][j] = __builtin_amdgcn_wmma_f32_16x16x32_bf16(
            false, aL[i], false, bH[j], (short)0, acc[i][j], false, false);
      }
    }

    __builtin_amdgcn_s_wait_tensorcnt(0);   // wave0 drains next-buffer DMA
    __syncthreads();                        // publish to all waves
  }

  // epilogue: C/D layout -> lanes 0-15: M=r, lanes 16-31: M=r+8; N = lane&15
  const int nl   = lane & 15;
  const int mOff = (lane >> 4) * 8;
  #pragma unroll
  for (int i = 0; i < 2; ++i) {
    #pragma unroll
    for (int j = 0; j < 2; ++j) {
      int col  = bn + wn * 32 + j * 16 + nl;
      int row0 = bm + wm * 32 + i * 16 + mOff;
      float bv = bias[col];
      #pragma unroll
      for (int r = 0; r < 8; ++r)
        Y[(size_t)(row0 + r) * N + col] = acc[i][j][r] + bv;
    }
  }
}

// ---------------- KNN: wave per node, lane-local top-7 + wave-min merge ----------------
__global__ __launch_bounds__(256) void knn_kernel(const float2* __restrict__ pos,
                                                  int* __restrict__ nbr)
{
  int gw   = (int)((blockIdx.x * blockDim.x + threadIdx.x) >> 5);
  int lane = threadIdx.x & 31;
  if (gw >= N_NODES) return;
  const int i = gw;
  float2 pi = pos[i];
  float sqi = pi.x * pi.x + pi.y * pi.y;

  float bd[K_NBR + 1];
  int   bi[K_NBR + 1];
  #pragma unroll
  for (int t = 0; t <= K_NBR; ++t) { bd[t] = 3.4e38f; bi[t] = 0x7FFFFFFF; }

  for (int j = lane; j < N_NODES; j += 32) {
    float2 pj = pos[j];
    float sqj = pj.x * pj.x + pj.y * pj.y;
    float d2 = sqi + sqj - 2.0f * (pi.x * pj.x + pi.y * pj.y);
    if (d2 < bd[K_NBR]) {       // strict <: ascending j keeps (d, idx) order stable
      float dc = d2; int ic = j;
      #pragma unroll
      for (int t = 0; t <= K_NBR; ++t) {
        if (dc < bd[t]) {
          float td = bd[t]; int ti = bi[t];
          bd[t] = dc; bi[t] = ic; dc = td; ic = ti;
        }
      }
    }
  }

  // 7 rounds of wave-wide arg-min over each lane's sorted list head
  int head = 0;
  #pragma unroll
  for (int r = 0; r <= K_NBR; ++r) {
    float mv = bd[head]; int mi = bi[head]; int ml = lane;
    #pragma unroll
    for (int o = 16; o > 0; o >>= 1) {
      float ov = __shfl_xor(mv, o, 32);
      int   oi = __shfl_xor(mi, o, 32);
      int   ol = __shfl_xor(ml, o, 32);
      if (ov < mv || (ov == mv && oi < mi)) { mv = ov; mi = oi; ml = ol; }
    }
    if (r >= 1 && lane == 0) nbr[i * K_NBR + (r - 1)] = mi;   // r=0 is self
    if (lane == ml) head++;
  }
}

// ---------------- edge helpers ----------------
static __device__ inline unsigned encode_f(float f) {
  unsigned u = __float_as_uint(f);
  return (u & 0x80000000u) ? ~u : (u | 0x80000000u);
}
static __device__ inline float decode_f(unsigned u) {
  unsigned b = (u & 0x80000000u) ? (u & 0x7FFFFFFFu) : ~u;
  return __uint_as_float(b);
}
static __device__ inline void edge_nodes(int eid, const int* __restrict__ nbr,
                                         int& src, int& dst) {
  if (eid < N_NODES * K_NBR) { src = eid / K_NBR; dst = nbr[eid]; }
  else                       { src = dst = eid - N_NODES * K_NBR; }  // self loop
}

// ---------------- e = sum(leaky_relu(xr[dst]+xl[src]) * att); segment max ----------------
__global__ __launch_bounds__(256) void edge_e_kernel(
    const float* __restrict__ xl, const float* __restrict__ xr,
    const float* __restrict__ att, const int* __restrict__ nbr,
    float* __restrict__ ebuf, unsigned int* __restrict__ mEnc)
{
  int gw   = (int)((blockIdx.x * blockDim.x + threadIdx.x) >> 5);
  int lane = threadIdx.x & 31;
  if (gw >= N_EDGES * H_HEADS) return;
  int eid = gw / H_HEADS, h = gw - eid * H_HEADS;
  int src, dst; edge_nodes(eid, nbr, src, dst);
  const float4* pl = (const float4*)(xl + (size_t)src * HC + h * C_CH);
  const float4* pr = (const float4*)(xr + (size_t)dst * HC + h * C_CH);
  const float4* pa = (const float4*)(att + h * C_CH);
  float s = 0.f;
  #pragma unroll
  for (int t = 0; t < C_CH / (4 * 32); ++t) {
    int c = t * 32 + lane;
    float4 a = pl[c], b = pr[c], w = pa[c];
    float v;
    v = a.x + b.x; v = (v >= 0.f) ? v : NEG_SLOPE * v; s = fmaf(v, w.x, s);
    v = a.y + b.y; v = (v >= 0.f) ? v : NEG_SLOPE * v; s = fmaf(v, w.y, s);
    v = a.z + b.z; v = (v >= 0.f) ? v : NEG_SLOPE * v; s = fmaf(v, w.z, s);
    v = a.w + b.w; v = (v >= 0.f) ? v : NEG_SLOPE * v; s = fmaf(v, w.w, s);
  }
  #pragma unroll
  for (int o = 16; o > 0; o >>= 1) s += __shfl_xor(s, o, 32);
  if (lane == 0) {
    ebuf[gw] = s;
    atomicMax(&mEnc[dst * H_HEADS + h], encode_f(s));
  }
}

// ---------------- ex = exp(e - m[dst]); segment sum ----------------
__global__ void edge_exp_kernel(const float* __restrict__ ebuf,
                                const unsigned int* __restrict__ mEnc,
                                const int* __restrict__ nbr,
                                float* __restrict__ exbuf, float* __restrict__ ssum)
{
  int g = blockIdx.x * blockDim.x + threadIdx.x;
  if (g >= N_EDGES * H_HEADS) return;
  int eid = g / H_HEADS, h = g - eid * H_HEADS;
  int src, dst; edge_nodes(eid, nbr, src, dst);
  (void)src;
  float m  = decode_f(mEnc[dst * H_HEADS + h]);
  float ex = __expf(ebuf[g] - m);
  exbuf[g] = ex;
  atomicAdd(&ssum[dst * H_HEADS + h], ex);
}

// ---------------- out[dst] += (alpha/H) * xl[src]  (head-mean fused) ----------------
__global__ __launch_bounds__(256) void edge_scatter_kernel(
    const float* __restrict__ xl, const float* __restrict__ exbuf,
    const float* __restrict__ ssum, const int* __restrict__ nbr,
    float* __restrict__ accb)
{
  int gw   = (int)((blockIdx.x * blockDim.x + threadIdx.x) >> 5);
  int lane = threadIdx.x & 31;
  if (gw >= N_EDGES * H_HEADS) return;
  int eid = gw / H_HEADS, h = gw - eid * H_HEADS;
  int src, dst; edge_nodes(eid, nbr, src, dst);
  float alpha = exbuf[gw] / (ssum[dst * H_HEADS + h] + 1e-16f);
  float scale = alpha * (1.0f / (float)H_HEADS);
  const float4* pl = (const float4*)(xl + (size_t)src * HC + h * C_CH);
  float* pd = accb + (size_t)dst * C_CH;
  #pragma unroll
  for (int t = 0; t < C_CH / (4 * 32); ++t) {
    int c = t * 32 + lane;
    float4 v = pl[c];
    atomicAdd(&pd[4 * c + 0], scale * v.x);
    atomicAdd(&pd[4 * c + 1], scale * v.y);
    atomicAdd(&pd[4 * c + 2], scale * v.z);
    atomicAdd(&pd[4 * c + 3], scale * v.w);
  }
}

// ---------------- final FC: out = (acc + gat_bias) @ fc_W + fc_b ----------------
__global__ void fc_kernel(const float* __restrict__ accb, const float* __restrict__ gbias,
                          const float* __restrict__ fcW, const float* __restrict__ fcb,
                          float* __restrict__ out)
{
  int gw   = (int)((blockIdx.x * blockDim.x + threadIdx.x) >> 5);
  int lane = threadIdx.x & 31;
  if (gw >= N_NODES) return;
  const float4* pa = (const float4*)(accb + (size_t)gw * C_CH);
  const float4* pg = (const float4*)gbias;
  const float4* pw = (const float4*)fcW;
  float s0 = 0.f, s1 = 0.f;
  #pragma unroll
  for (int t = 0; t < C_CH / (4 * 32); ++t) {
    int c = t * 32 + lane;
    float4 a = pa[c], g = pg[c];
    float4 w0 = pw[2 * c], w1 = pw[2 * c + 1];
    float v;
    v = a.x + g.x; s0 = fmaf(v, w0.x, s0); s1 = fmaf(v, w0.y, s1);
    v = a.y + g.y; s0 = fmaf(v, w0.z, s0); s1 = fmaf(v, w0.w, s1);
    v = a.z + g.z; s0 = fmaf(v, w1.x, s0); s1 = fmaf(v, w1.y, s1);
    v = a.w + g.w; s0 = fmaf(v, w1.z, s0); s1 = fmaf(v, w1.w, s1);
  }
  #pragma unroll
  for (int o = 16; o > 0; o >>= 1) {
    s0 += __shfl_xor(s0, o, 32);
    s1 += __shfl_xor(s1, o, 32);
  }
  if (lane == 0) {
    out[2 * gw + 0] = s0 + fcb[0];
    out[2 * gw + 1] = s1 + fcb[1];
  }
}

// ---------------- host launcher ----------------
extern "C" void kernel_launch(void* const* d_in, const int* in_sizes, int n_in,
                              void* d_out, int out_size, void* d_ws, size_t ws_size,
                              hipStream_t stream)
{
  (void)in_sizes; (void)n_in; (void)out_size; (void)ws_size;

  const float* feature  = (const float*)d_in[0];  // (1,N,D)
  const float* position = (const float*)d_in[1];  // (N,2)
  const float* W_l      = (const float*)d_in[2];  // (D, H*C)
  const float* b_l      = (const float*)d_in[3];  // (H*C)
  const float* W_r      = (const float*)d_in[4];
  const float* b_r      = (const float*)d_in[5];
  const float* att      = (const float*)d_in[6];  // (H,C)
  const float* gat_bias = (const float*)d_in[7];  // (C)
  const float* fc_W     = (const float*)d_in[8];  // (C,2)
  const float* fc_b     = (const float*)d_in[9];  // (2)
  float*       out      = (float*)d_out;          // (N,2)

  char* ws = (char*)d_ws;
  size_t off = 0;
  auto take = [&](size_t bytes) -> void* {
    off = (off + 255) & ~(size_t)255;
    void* p = ws + off;
    off += bytes;
    return p;
  };
  int*            nbr   = (int*)take(sizeof(int) * (size_t)N_NODES * K_NBR);
  float*          xl    = (float*)take(sizeof(float) * (size_t)N_NODES * HC);
  float*          xr    = (float*)take(sizeof(float) * (size_t)N_NODES * HC);
  float*          ebuf  = (float*)take(sizeof(float) * (size_t)N_EDGES * H_HEADS);
  float*          exbuf = (float*)take(sizeof(float) * (size_t)N_EDGES * H_HEADS);
  unsigned int*   mEnc  = (unsigned int*)take(sizeof(unsigned) * (size_t)N_NODES * H_HEADS);
  float*          ssum  = (float*)take(sizeof(float) * (size_t)N_NODES * H_HEADS);
  float*          accb  = (float*)take(sizeof(float) * (size_t)N_NODES * C_CH);
  unsigned short* xhi   = (unsigned short*)take(sizeof(short) * (size_t)N_NODES * D_IN);
  unsigned short* xlo   = (unsigned short*)take(sizeof(short) * (size_t)N_NODES * D_IN);
  unsigned short* wlThi = (unsigned short*)take(sizeof(short) * (size_t)HC * D_IN);
  unsigned short* wlTlo = (unsigned short*)take(sizeof(short) * (size_t)HC * D_IN);
  unsigned short* wrThi = (unsigned short*)take(sizeof(short) * (size_t)HC * D_IN);
  unsigned short* wrTlo = (unsigned short*)take(sizeof(short) * (size_t)HC * D_IN);

  // 1) KNN graph (wave per node)
  knn_kernel<<<((size_t)N_NODES * 32 + 255) / 256, 256, 0, stream>>>(
      (const float2*)position, nbr);

  // 2) precision-split conversions (bandwidth-trivial, L2-resident)
  const int xN4 = (N_NODES * D_IN) / 4;
  cvt_split_kernel<<<(xN4 + 255) / 256, 256, 0, stream>>>(feature, xhi, xlo, xN4);
  cvt_wT_kernel<<<(HC * D_IN + 255) / 256, 256, 0, stream>>>(W_l, wlThi, wlTlo);
  cvt_wT_kernel<<<(HC * D_IN + 255) / 256, 256, 0, stream>>>(W_r, wrThi, wrTlo);

  // 3) WMMA GEMMs, TDM-staged + double-buffered
  dim3 gg(N_NODES / TM, HC / TN, 1);  // 125 x 12
  gemm_bias_bf16split_tdm<<<gg, 256, 0, stream>>>(xhi, xlo, wlThi, wlTlo, b_l, xl,
                                                  N_NODES, D_IN, HC);
  gemm_bias_bf16split_tdm<<<gg, 256, 0, stream>>>(xhi, xlo, wrThi, wrTlo, b_r, xr,
                                                  N_NODES, D_IN, HC);

  // 4) zero segment buffers (graph-capture-safe async memsets)
  hipMemsetAsync(mEnc, 0, sizeof(unsigned) * (size_t)N_NODES * H_HEADS, stream);
  hipMemsetAsync(ssum, 0, sizeof(float) * (size_t)N_NODES * H_HEADS, stream);
  hipMemsetAsync(accb, 0, sizeof(float) * (size_t)N_NODES * C_CH, stream);

  // 5) GATv2 edge phase
  const int totalEH = N_EDGES * H_HEADS;  // 168000
  edge_e_kernel<<<((size_t)totalEH * 32 + 255) / 256, 256, 0, stream>>>(
      xl, xr, att, nbr, ebuf, mEnc);
  edge_exp_kernel<<<(totalEH + 255) / 256, 256, 0, stream>>>(
      ebuf, mEnc, nbr, exbuf, ssum);
  edge_scatter_kernel<<<((size_t)totalEH * 32 + 255) / 256, 256, 0, stream>>>(
      xl, exbuf, ssum, nbr, accb);

  // 6) classifier head
  fc_kernel<<<((size_t)N_NODES * 32 + 255) / 256, 256, 0, stream>>>(
      accb, gat_bias, fc_W, fc_b, out);
}
